// GaussianMixtureModel_8229157339640
// MI455X (gfx1250) — matrix operand order repaired and checked by reading, not compiled, not run on hardware
//
#include <hip/hip_runtime.h>
#include <math.h>

#define GMM_N 200000
#define GMM_K 16
#define GMM_D 64
#define ROWS_PER_BLOCK 256                 // 8 waves x 2 tiles x 16 rows
#define NBLOCKS ((GMM_N + ROWS_PER_BLOCK - 1) / ROWS_PER_BLOCK)  // 782

typedef __attribute__((ext_vector_type(8)))  float  v8f;
typedef __attribute__((ext_vector_type(16))) __bf16 v16bf;

#define WMMA_BF16(A, B, C) \
  __builtin_amdgcn_wmma_f32_16x16x32_bf16(false, (A), false, (B), (short)0, (C), false, false)

// ---------------------------------------------------------------------------
// hi/lo bf16 split helpers (prep-side packing)
// ---------------------------------------------------------------------------
__device__ inline unsigned short bf16bits(float x) {
  __bf16 b = (__bf16)x;
  return __builtin_bit_cast(unsigned short, b);
}
__device__ inline float bf16val(float x) { return (float)(__bf16)x; }

__device__ inline unsigned pack_rep(float a, float b, int rep) {
  float ar = rep ? (a - bf16val(a)) : a;
  float br = rep ? (b - bf16val(b)) : b;
  return (unsigned)bf16bits(ar) | ((unsigned)bf16bits(br) << 16);
}

// ---------------------------------------------------------------------------
// Prep kernel, one block per component k (64 threads):
//   Cholesky L of cov; W = L^{-1}; b = W mu; v = W^T b = Sigma^{-1} mu;
//   c'_k = log pi_k - sum log L_ii - 32 log 2pi - 0.5 ||b||^2
//   Emit Whi/Wlo and vhat hi/lo pre-swizzled into the bf16 WMMA A-operand
//   layout (lane = (h<<4)|m holds row M=m; VGPR v packs
//   Klocal = 16*(v>>2) + 8*h + 2*(v&3) + {0,1}).
// ---------------------------------------------------------------------------
__global__ __launch_bounds__(64) void gmm_prep(
    const float* __restrict__ pi, const float* __restrict__ mus,
    const float* __restrict__ covs, unsigned* __restrict__ gA,
    unsigned* __restrict__ gV, float* __restrict__ gc)
{
  __shared__ float Asm[64 * 65];
  __shared__ float Wsm[64 * 65];
  __shared__ float musm[64];
  __shared__ float bsm[64];
  __shared__ float vsm[64];

  const int k = blockIdx.x;
  const int tid = threadIdx.x;

  for (int j = 0; j < 64; ++j)
    Asm[tid * 65 + j] = covs[k * 4096 + tid * 64 + j];
  musm[tid] = mus[k * 64 + tid];
  __syncthreads();

  // Right-looking Cholesky, lower triangle in place.
  for (int j = 0; j < 64; ++j) {
    if (tid == 0) Asm[j * 65 + j] = sqrtf(Asm[j * 65 + j]);
    __syncthreads();
    const float djj = Asm[j * 65 + j];
    if (tid > j) Asm[tid * 65 + j] /= djj;
    __syncthreads();
    if (tid > j) {
      const float lij = Asm[tid * 65 + j];
      for (int c2 = j + 1; c2 <= tid; ++c2)
        Asm[tid * 65 + c2] -= lij * Asm[c2 * 65 + j];
    }
    __syncthreads();
  }

  // W = L^{-1}; thread tid owns column tid (upper part zeroed).
  {
    const int c = tid;
    for (int i = 0; i < 64; ++i) {
      float w = 0.f;
      if (i >= c) {
        w = (i == c) ? 1.f : 0.f;
        for (int p = c; p < i; ++p) w -= Asm[i * 65 + p] * Wsm[p * 65 + c];
        w /= Asm[i * 65 + i];
      }
      Wsm[i * 65 + c] = w;
    }
  }
  __syncthreads();

  // b = W mu (row tid)
  {
    float bb = 0.f;
    for (int d = 0; d <= tid; ++d) bb += Wsm[tid * 65 + d] * musm[d];
    bsm[tid] = bb;
  }
  __syncthreads();

  // v = W^T b (entry tid)
  {
    float vv = 0.f;
    for (int i = tid; i < 64; ++i) vv += Wsm[i * 65 + tid] * bsm[i];
    vsm[tid] = vv;
  }

  if (tid == 0) {
    float hld = 0.f, b2 = 0.f;
    for (int d = 0; d < 64; ++d) {
      hld += logf(Asm[d * 65 + d]);
      b2 += bsm[d] * bsm[d];
    }
    gc[k] = logf(pi[k]) - hld - 58.81206612509905f - 0.5f * b2;
  }
  __syncthreads();

  // W A-operand swizzle: 4096 u32 per component.
  // e bits: t(2) | kk(1) | rep(1) | lane(5) | v(3)
  for (int e = tid; e < 4096; e += 64) {
    const int v    = e & 7;
    const int lane = (e >> 3) & 31;
    const int rep  = (e >> 8) & 1;
    const int kk   = (e >> 9) & 1;
    const int t    = (e >> 10) & 3;
    const int m = lane & 15, h = lane >> 4;
    const int i = 16 * t + m;
    const int d0 = 32 * kk + 16 * (v >> 2) + 8 * h + 2 * (v & 3);
    gA[k * 4096 + e] = pack_rep(Wsm[i * 65 + d0], Wsm[i * 65 + d0 + 1], rep);
  }

  // vhat A-operand swizzle: this block fills the 64 entries where M == k.
  {
    const int v   = tid & 7;
    const int h   = (tid >> 3) & 1;
    const int rep = (tid >> 4) & 1;
    const int kk  = (tid >> 5) & 1;
    const int lane = (h << 4) | k;
    const int e = (kk << 9) | (rep << 8) | (lane << 3) | v;
    const int d0 = 32 * kk + 16 * (v >> 2) + 8 * h + 2 * (v & 3);
    gV[e] = pack_rep(vsm[d0], vsm[d0 + 1], rep);
  }
}

// ---------------------------------------------------------------------------
// Main kernel: 8 waves x 32 rows (2 B-tiles per wave). Whi/Wlo (256 KB) +
// vhat (4 KB) in LDS. Each W operand fetched from LDS is reused against BOTH
// X tiles -> ~341 B of DS traffic per wmma, within LDS bandwidth.
// ---------------------------------------------------------------------------
__global__ __launch_bounds__(256, 1) void gmm_main(
    const float* __restrict__ X, const unsigned* __restrict__ gA,
    const unsigned* __restrict__ gV, const float* __restrict__ gc,
    float* __restrict__ out)
{
  extern __shared__ unsigned smem[];
  unsigned* sA = smem;                    // 65536 u32 : W operands, swizzled
  unsigned* sV = smem + 65536;            // 1024 u32  : vhat operands
  float* scc   = (float*)(smem + 66560);  // 16 floats : c'_k
  float* slin  = (float*)(smem + 66576);  // 8*2*16*16 : Lin/weighted [w][t][k][r]

  // Cooperative LDS fill.
  {
    const uint4* srcA = (const uint4*)gA;
    uint4* dstA = (uint4*)sA;
    for (int i = threadIdx.x; i < 16384; i += 256) dstA[i] = srcA[i];
    const uint4* srcV = (const uint4*)gV;
    uint4* dstV = (uint4*)sV;
    for (int i = threadIdx.x; i < 256; i += 256) dstV[i] = srcV[i];
    if (threadIdx.x < 16) scc[threadIdx.x] = gc[threadIdx.x];
  }
  __syncthreads();

  const int wave = threadIdx.x >> 5;
  const int lane = threadIdx.x & 31;
  const int hf   = lane >> 4;
  const int n    = lane & 15;
  const v16bf* pA = (const v16bf*)sA;   // ((k*4+t)*2+kk)*2+rep)*32+lane
  const v16bf* pV = (const v16bf*)sV;   // (kk*2+rep)*32+lane
  float* mylin = slin + wave * 512;     // [tile][comp][row]

  for (int blk = blockIdx.x; blk < NBLOCKS; blk += gridDim.x) {
    const int rowBase = blk * ROWS_PER_BLOCK + wave * 32;
    const bool active = (rowBase + 32) <= GMM_N;  // wave-uniform (N % 32 == 0)

    if (active) {
      // ---- X tiles -> bf16 hi/lo B operands ------------------------------
      // B layout: lane holds row n; VGPR v packs K dims
      // d = 32*kk + 16*(v>>2) + 8*hf + 2*(v&3) + {0,1}.  xh/xl index: [tile][kk]
      v16bf xh[2][2], xl[2][2];
      const float* xb = X + (size_t)(rowBase + n) * GMM_D + 8 * hf;
#pragma unroll
      for (int ti = 0; ti < 2; ++ti) {
#pragma unroll
        for (int kk = 0; kk < 2; ++kk) {
#pragma unroll
          for (int v = 0; v < 8; ++v) {
            const float x0 = xb[ti * 1024 + kk * 32 + 16 * (v >> 2) + 2 * (v & 3)];
            const float x1 = xb[ti * 1024 + kk * 32 + 16 * (v >> 2) + 2 * (v & 3) + 1];
            const __bf16 h0 = (__bf16)x0, h1 = (__bf16)x1;
            xh[ti][kk][2 * v]     = h0;
            xh[ti][kk][2 * v + 1] = h1;
            xl[ti][kk][2 * v]     = (__bf16)(x0 - (float)h0);
            xl[ti][kk][2 * v + 1] = (__bf16)(x1 - (float)h1);
          }
        }
      }

      // ---- Lin = vhat * X  (C: M=component, N=row) -----------------------
      {
        v8f accL0 = {}, accL1 = {};
#pragma unroll
        for (int kk = 0; kk < 2; ++kk) {
          const v16bf ah = pV[(kk * 2 + 0) * 32 + lane];
          const v16bf al = pV[(kk * 2 + 1) * 32 + lane];
          accL0 = WMMA_BF16(ah, xh[0][kk], accL0);
          accL1 = WMMA_BF16(ah, xh[1][kk], accL1);
          accL0 = WMMA_BF16(ah, xl[0][kk], accL0);
          accL1 = WMMA_BF16(ah, xl[1][kk], accL1);
          accL0 = WMMA_BF16(al, xh[0][kk], accL0);
          accL1 = WMMA_BF16(al, xh[1][kk], accL1);
        }
#pragma unroll
        for (int v = 0; v < 8; ++v) {
          mylin[(v + 8 * hf) * 16 + n]       = accL0[v];
          mylin[256 + (v + 8 * hf) * 16 + n] = accL1[v];
        }
      }

      // ---- per-component whitened GEMM + maha ----------------------------
      for (int k = 0; k < GMM_K; ++k) {
        v8f a00 = {}, a10 = {}, a20 = {}, a30 = {};  // tile 0, 4 dim-tiles
        v8f a01 = {}, a11 = {}, a21 = {}, a31 = {};  // tile 1
        const v16bf* pAk = pA + k * 512;
#pragma unroll
        for (int kk = 0; kk < 2; ++kk) {
          // W-hi operands for the 4 output-dim tiles
          const v16bf w0 = pAk[((0 * 2 + kk) * 2 + 0) * 32 + lane];
          const v16bf w1 = pAk[((1 * 2 + kk) * 2 + 0) * 32 + lane];
          const v16bf w2 = pAk[((2 * 2 + kk) * 2 + 0) * 32 + lane];
          const v16bf w3 = pAk[((3 * 2 + kk) * 2 + 0) * 32 + lane];
          a00 = WMMA_BF16(w0, xh[0][kk], a00);
          a10 = WMMA_BF16(w1, xh[0][kk], a10);
          a20 = WMMA_BF16(w2, xh[0][kk], a20);
          a30 = WMMA_BF16(w3, xh[0][kk], a30);
          a01 = WMMA_BF16(w0, xh[1][kk], a01);
          a11 = WMMA_BF16(w1, xh[1][kk], a11);
          a21 = WMMA_BF16(w2, xh[1][kk], a21);
          a31 = WMMA_BF16(w3, xh[1][kk], a31);
          a00 = WMMA_BF16(w0, xl[0][kk], a00);
          a10 = WMMA_BF16(w1, xl[0][kk], a10);
          a20 = WMMA_BF16(w2, xl[0][kk], a20);
          a30 = WMMA_BF16(w3, xl[0][kk], a30);
          a01 = WMMA_BF16(w0, xl[1][kk], a01);
          a11 = WMMA_BF16(w1, xl[1][kk], a11);
          a21 = WMMA_BF16(w2, xl[1][kk], a21);
          a31 = WMMA_BF16(w3, xl[1][kk], a31);
          // W-lo operands (reuse register space after hi is consumed)
          const v16bf u0 = pAk[((0 * 2 + kk) * 2 + 1) * 32 + lane];
          const v16bf u1 = pAk[((1 * 2 + kk) * 2 + 1) * 32 + lane];
          const v16bf u2 = pAk[((2 * 2 + kk) * 2 + 1) * 32 + lane];
          const v16bf u3 = pAk[((3 * 2 + kk) * 2 + 1) * 32 + lane];
          a00 = WMMA_BF16(u0, xh[0][kk], a00);
          a10 = WMMA_BF16(u1, xh[0][kk], a10);
          a20 = WMMA_BF16(u2, xh[0][kk], a20);
          a30 = WMMA_BF16(u3, xh[0][kk], a30);
          a01 = WMMA_BF16(u0, xh[1][kk], a01);
          a11 = WMMA_BF16(u1, xh[1][kk], a11);
          a21 = WMMA_BF16(u2, xh[1][kk], a21);
          a31 = WMMA_BF16(u3, xh[1][kk], a31);
        }
        // maha: in-lane sum over 32 dims (v x t) per tile, then fold halves.
        float p0 = 0.f, p1 = 0.f;
#pragma unroll
        for (int v = 0; v < 8; ++v) {
          p0 += a00[v] * a00[v];
          p0 += a10[v] * a10[v];
          p0 += a20[v] * a20[v];
          p0 += a30[v] * a30[v];
          p1 += a01[v] * a01[v];
          p1 += a11[v] * a11[v];
          p1 += a21[v] * a21[v];
          p1 += a31[v] * a31[v];
        }
        p0 += __shfl_xor(p0, 16, 32);
        p1 += __shfl_xor(p1, 16, 32);
        // lanes 0-15 finalize tile0's row n; lanes 16-31 tile1's row n
        {
          const float p = hf ? p1 : p0;
          float* lp = mylin + hf * 256 + k * 16 + n;
          const float lin = *lp;
          *lp = scc[k] + lin - 0.5f * p;
        }
      }
    }
    __syncthreads();

    // ---- softmax over K per row (lane L owns row 16*hf + n) --------------
    if (active) {
      const float* lp = mylin + hf * 256;
      float w[16];
      float mx = -1e30f;
#pragma unroll
      for (int k = 0; k < 16; ++k) {
        w[k] = lp[k * 16 + n];
        mx = fmaxf(mx, w[k]);
      }
      float s = 0.f;
#pragma unroll
      for (int k = 0; k < 16; ++k) {
        w[k] = __expf(w[k] - mx);
        s += w[k];
      }
      const float inv = 1.f / s;
      float4* o = (float4*)(out + (size_t)(rowBase + 16 * hf + n) * GMM_K);
      o[0] = make_float4(w[0] * inv, w[1] * inv, w[2] * inv, w[3] * inv);
      o[1] = make_float4(w[4] * inv, w[5] * inv, w[6] * inv, w[7] * inv);
      o[2] = make_float4(w[8] * inv, w[9] * inv, w[10] * inv, w[11] * inv);
      o[3] = make_float4(w[12] * inv, w[13] * inv, w[14] * inv, w[15] * inv);
    }
    __syncthreads();
  }
}

// ---------------------------------------------------------------------------
extern "C" void kernel_launch(void* const* d_in, const int* in_sizes, int n_in,
                              void* d_out, int out_size, void* d_ws, size_t ws_size,
                              hipStream_t stream) {
  (void)in_sizes; (void)n_in; (void)out_size; (void)ws_size;
  const float* X    = (const float*)d_in[0];   // [N, 64]
  const float* pi   = (const float*)d_in[1];   // [16]
  const float* mus  = (const float*)d_in[2];   // [16, 64]
  const float* covs = (const float*)d_in[3];   // [16, 64, 64]
  float* out = (float*)d_out;                  // [N, 16]

  unsigned* ws = (unsigned*)d_ws;              // needs ~266 KB
  unsigned* gA = ws;                           // 16*4096 u32 (Whi/Wlo swizzled)
  unsigned* gV = ws + 16 * 4096;               // 1024 u32 (vhat swizzled)
  float*    gc = (float*)(ws + 16 * 4096 + 1024);  // 16 floats

  gmm_prep<<<GMM_K, 64, 0, stream>>>(pi, mus, covs, gA, gV, gc);

  const size_t shmem = (size_t)(65536 + 1024 + 16 + 8 * 2 * 16 * 16) * 4;  // 282688
  hipFuncSetAttribute((const void*)gmm_main,
                      hipFuncAttributeMaxDynamicSharedMemorySize, (int)shmem);
  gmm_main<<<NBLOCKS, 256, shmem, stream>>>(X, gA, gV, gc, out);
}
